// MultiHeadedAttention_42700564857257
// MI455X (gfx1250) — matrix operand order repaired and checked by reading, not compiled, not run on hardware
//
#include <hip/hip_runtime.h>

typedef __bf16 bf16_t;
typedef __attribute__((ext_vector_type(8)))  __bf16 v8bf;
typedef __attribute__((ext_vector_type(16))) __bf16 v16bf;
typedef __attribute__((ext_vector_type(8)))  float  v8f;
typedef __attribute__((ext_vector_type(4)))  unsigned int v4u;
typedef __attribute__((ext_vector_type(8)))  int v8i;
typedef __attribute__((ext_vector_type(4)))  int v4i;

// Problem constants (match reference)
#define HH   8
#define DKK  64
#define NBB  4
#define DMM  512
#define CCC  17
#define NNN  1024
#define BBB  2

// ---------------- fragment helpers ----------------
// A fragment 16x32 bf16: lane holds row m=lane&15; lanes 0-15 hold K=0..15,
// lanes 16-31 hold K=16..31 (one contiguous 16-elem run, two 8-elem loads).
__device__ __forceinline__ v16bf cat8(v8bf lo, v8bf hi) {
  return __builtin_shufflevector(lo, hi, 0,1,2,3,4,5,6,7,8,9,10,11,12,13,14,15);
}
__device__ __forceinline__ v16bf load_a_frag(const bf16_t* p, int ld, int lane) {
  int m = lane & 15;
  int s = (lane >> 4) << 4;
  const bf16_t* r = p + (size_t)m * ld + s;
  v8bf lo = *(const v8bf*)(r);
  v8bf hi = *(const v8bf*)(r + 8);
  return cat8(lo, hi);
}
// B fragment 32x16 bf16: lane holds column n=lane&15; lanes 0-15 hold K=0..15,
// lanes 16-31 hold K=16..31 (one contiguous 16-elem run). Memory layout must be
// [col][k] row-major (column's k-run contiguous).
__device__ __forceinline__ v16bf load_b_frag(const bf16_t* p, int ld, int lane) {
  int n = lane & 15;
  int s = (lane >> 4) << 4;
  return *(const v16bf*)(p + (size_t)n * ld + s);
}
__device__ __forceinline__ v8f wmma_bf16(v16bf a, v16bf b, v8f c) {
  return __builtin_amdgcn_wmma_f32_16x16x32_bf16(false, a, false, b, (short)0, c,
                                                 false, false);
}

// ---------------- TDM: issue one tensor_load_to_lds -------------------------
// D# group0: count=1 (user), lds_addr, 57-bit global addr, type=2 ("image").
__device__ __forceinline__ void tdm_issue(unsigned lds_off, const bf16_t* g,
                                          v8i g1) {
  unsigned long long ga = (unsigned long long)(size_t)g;
  v4u g0;
  g0[0] = 1u;                                   // count=1, is_restore=0
  g0[1] = lds_off;                              // LDS byte address
  g0[2] = (unsigned)ga;                         // global_addr[31:0]
  g0[3] = (unsigned)((ga >> 32) & 0x01FFFFFFu)  // global_addr[56:32]
          | 0x80000000u;                        // type=2 (bits 127:126 = 10b)
  v4i z4 = {0, 0, 0, 0};
#if __clang_major__ >= 23
  v8i z8 = {0, 0, 0, 0, 0, 0, 0, 0};
  __builtin_amdgcn_tensor_load_to_lds(g0, g1, z4, z4, z8, 0);
#else
  __builtin_amdgcn_tensor_load_to_lds(g0, g1, z4, z4, 0);
#endif
}

// Stage 5 K-bank tiles (32x64 bf16, contiguous 4KB each) and 5 V'^T-bank tiles
// (64 rows x 32 cols bf16, row stride 1024 elems) for one jb into an LDS slot.
__device__ __forceinline__ void stage_tiles(unsigned smem0, int slot,
                                            const bf16_t* Kbank,
                                            const bf16_t* VbankT, int bh,
                                            int jb) {
  // group1 for K: data_size=2B, tensor_dim0=2048, tensor_dim1=1,
  //               tile_dim0=2048 (1-D contiguous copy), stride0=2048.
  const v8i g1K = {(int)(1u << 16), 0x08000000, 0x00010000, 0x08000000,
                   0, 2048, 0, 0};
  // group1 for V: data_size=2B, tensor_dim0=1024, tensor_dim1=64,
  //               tile_dim0=32, tile_dim1=64, stride0=1024.
  const v8i g1V = {(int)(1u << 16), 0x04000000, 0x00400000, 0x00200000,
                   64, 1024, 0, 0};
  unsigned so = smem0 + (unsigned)slot * 40960u;
#pragma unroll
  for (int bi = 0; bi < 5; ++bi) {
    tdm_issue(so + (unsigned)bi * 4096u,
              Kbank + (((size_t)(bi * 16 + bh)) * 1024 + jb) * 64, g1K);
    tdm_issue(so + 20480u + (unsigned)bi * 4096u,
              VbankT + (((size_t)(bi * 16 + bh)) * 64) * 1024 + jb, g1V);
  }
}

// ---------------- 1) transpose projection weights to bf16 [s*5][out][d] ----
__global__ void moba_prep_weights(const float* __restrict__ lin,
                                  bf16_t* __restrict__ Wt) {
  int t = blockIdx.x * blockDim.x + threadIdx.x;
  if (t >= 3 * 5 * 512 * 512) return;
  int d  = t & 511;
  int o  = (t >> 9) & 511;
  int sb = t >> 18;
  // lin: [3][5][d=512][o=512] row-major ; Wt: [3*5][o][d]
  Wt[t] = (bf16_t)lin[((size_t)(sb * 512 + d)) * 512 + o];
}

// ---------------- 2) softmax-blend W1m [c][h][m][n], W2mT [c][h][D][d] ------
__global__ void moba_mix(const float* __restrict__ W1, const float* __restrict__ a1,
                         const float* __restrict__ W2, const float* __restrict__ a2,
                         bf16_t* __restrict__ W1m, bf16_t* __restrict__ W2mT) {
  const int total = CCC * HH * 64 * 64;
  int tid = blockIdx.x * blockDim.x + threadIdx.x;
  bool second = tid >= total;
  int t = second ? tid - total : tid;
  if (t >= total) return;
  int n = t & 63, m = (t >> 6) & 63, h = (t >> 12) & 7, c = t >> 15;
  const float* A = second ? a2 : a1;
  float lg[4], mx = -3e38f;
#pragma unroll
  for (int q = 0; q < 4; ++q) { lg[q] = A[(c * 4 + q) * 8 + h]; mx = fmaxf(mx, lg[q]); }
  float s = 0.f;
#pragma unroll
  for (int q = 0; q < 4; ++q) { lg[q] = __expf(lg[q] - mx); s += lg[q]; }
  float inv = 1.f / s, acc = 0.f;
  if (!second) {
#pragma unroll
    for (int q = 0; q < 4; ++q)
      acc += lg[q] * inv * W1[(((size_t)(q * 8 + h)) * 64 + m) * 64 + n];
    W1m[t] = (bf16_t)acc;              // [c][h][m][n]
  } else {
#pragma unroll
    for (int q = 0; q < 4; ++q)        // transposed: element (D=m, d=n)
      acc += lg[q] * inv * W2[(((size_t)(q * 8 + h)) * 64 + n) * 64 + m];
    W2mT[t] = (bf16_t)acc;             // [c][h][D][d]
  }
}

// ---------------- 3) QKV projection via masked WMMA -------------------------
// out[tok, o] = sum_d X[tok,d] * Wt[blk(tok)][o][d]; mask A rows per block.
__global__ __launch_bounds__(128) void moba_proj(
    const float* __restrict__ Q, const float* __restrict__ K,
    const float* __restrict__ V, const bf16_t* __restrict__ Wt,
    const int* __restrict__ b_seq, bf16_t* __restrict__ qO,
    bf16_t* __restrict__ kO, bf16_t* __restrict__ vO) {
  int lane = threadIdx.x & 31, wv = threadIdx.x >> 5;
  int ct = blockIdx.y * 4 + wv;
  int o0 = ct * 16;
  int t0 = blockIdx.x * 16;          // flat token base (b*1024+n)
  int src = blockIdx.z;
  const float* X = (src == 0) ? Q : (src == 1) ? K : V;
  bf16_t* O = (src == 0) ? qO : (src == 1) ? kO : vO;
  const bf16_t* W = Wt + (size_t)src * 5 * 512 * 512;
  int m = lane & 15;
  int blk_i = b_seq[t0 + m];         // block of this lane's token row
  v16bf zf = {};
  v8f acc = {};
  for (int dc = 0; dc < 512; dc += 32) {
    v16bf a;
    {
      int s = (lane >> 4) << 4;
      const float* r = X + ((size_t)(t0 + m)) * 512 + dc + s;
#pragma unroll
      for (int e = 0; e < 8; ++e) a[e] = (bf16_t)r[e];
#pragma unroll
      for (int e = 0; e < 8; ++e) a[8 + e] = (bf16_t)r[8 + e];
    }
#pragma unroll
    for (int blk = 0; blk < 5; ++blk) {
      v16bf am = (blk_i == blk) ? a : zf;
      v16bf bf = load_b_frag(W + ((size_t)(blk * 512 + o0)) * 512 + dc, 512, lane);
      acc = wmma_bf16(am, bf, acc);
    }
  }
  int o = o0 + (lane & 15);
  int h = o >> 6, kk = o & 63;
  int mo = (lane >> 4) << 3;
#pragma unroll
  for (int e = 0; e < 8; ++e) {
    int tok = t0 + mo + e;
    int b = tok >> 10, nn = tok & 1023;
    O[(((size_t)(b * 8 + h)) * 1024 + nn) * 64 + kk] = (bf16_t)acc[e];
  }
}

// ---------------- 4) K/V bank build (5 banks per row-block value) -----------
// Kbank[bi][bh][j][m] = sum_n k[j,n] W1m[c(bi,bj)][m][n]
// VbankT[bi][bh][D][j] = sum_d v[j,d] W2mT[c(bi,bj)][D][d]  (stored transposed)
__global__ __launch_bounds__(128) void moba_banks(
    const bf16_t* __restrict__ kB, const bf16_t* __restrict__ vB,
    const int* __restrict__ b_seq, const bf16_t* __restrict__ W1m,
    const bf16_t* __restrict__ W2mT, bf16_t* __restrict__ Kbank,
    bf16_t* __restrict__ VbankT) {
  int lane = threadIdx.x & 31, wv = threadIdx.x >> 5;
  int jt = blockIdx.x;
  int bi = blockIdx.y >> 4;
  int bh = blockIdx.y & 15;
  int kind = blockIdx.z;             // 0 = K banks, 1 = V banks
  int b = bh >> 3, h = bh & 7;
  int j0 = jt * 16;
  int c0 = wv * 16;
  const bf16_t* Src = kind ? vB : kB;
  const bf16_t* Wbase = kind ? W2mT : W1m;
  int m = lane & 15;
  int bj = b_seq[b * 1024 + j0 + m]; // block of this lane's key/value row
  v16bf zf = {};
  v8f acc = {};
#pragma unroll
  for (int nc = 0; nc < 64; nc += 32) {
    v16bf a = load_a_frag(Src + ((size_t)(bh * 1024 + j0)) * 64 + nc, 64, lane);
#pragma unroll
    for (int blk = 0; blk < 5; ++blk) {
      int cls = (bi == 0 || blk == 0) ? 0 : (bi - 1) * 4 + blk;
      const bf16_t* Wm = Wbase + ((size_t)(cls * 8 + h)) * 4096;
      v16bf am = (bj == blk) ? a : zf;
      v16bf bf = load_b_frag(Wm + c0 * 64 + nc, 64, lane);
      acc = wmma_bf16(am, bf, acc);
    }
  }
  int mo = (lane >> 4) << 3;
  if (kind == 0) {
    bf16_t* dst = Kbank + (((size_t)(bi * 16 + bh)) * 1024 + j0) * 64 + c0;
#pragma unroll
    for (int e = 0; e < 8; ++e)
      dst[(size_t)(mo + e) * 64 + (lane & 15)] = (bf16_t)acc[e];
  } else {
    bf16_t* dst = VbankT + (((size_t)(bi * 16 + bh)) * 64 + c0) * 1024 + j0;
#pragma unroll
    for (int e = 0; e < 8; ++e)
      dst[(size_t)(lane & 15) * 1024 + mo + e] = (bf16_t)acc[e];
  }
}

// ---------------- 5) flash attention with bank selection --------------------
// S^T = Kbank x Q^T  -> C lane = query column i (softmax stats per-lane).
// O   = V'^T x P^T   -> C lane = query column i (rescale per-lane).
// K/V bank tiles are staged into LDS once per workgroup by the Tensor Data
// Mover (double-buffered, TENSORcnt-tracked) and shared by all 4 waves.
__global__ __launch_bounds__(128) void moba_attn(
    const bf16_t* __restrict__ qB, const bf16_t* __restrict__ Kbank,
    const bf16_t* __restrict__ VbankT, const int* __restrict__ b_seq,
    const unsigned char* __restrict__ maskA, float* __restrict__ out) {
  extern __shared__ char smem_raw[];      // 2 slots x (20KB K + 20KB V)
  unsigned smem0 = (unsigned)(size_t)smem_raw;  // LDS byte offset for TDM D#
  int lane = threadIdx.x & 31, wv = threadIdx.x >> 5;
  int it = blockIdx.x * 4 + wv;
  int i0 = it * 16;
  int bh = blockIdx.y;
  int b = bh >> 3, h = bh & 7;
  int icol = i0 + (lane & 15);
  int bsel_i = b_seq[b * 1024 + icol];
  int mask_i = maskA[b * 1024 + icol];
  int hi8 = (lane >> 4) << 3;
  v16bf zf = {};
  const bf16_t* Qp = qB + ((size_t)bh * 1024) * 64;
  v16bf qf[2];
#pragma unroll
  for (int kc = 0; kc < 2; ++kc)
    qf[kc] = load_b_frag(Qp + (size_t)i0 * 64 + kc * 32, 64, lane);
  v8f O0 = {}, O1 = {}, O2 = {}, O3 = {};
  float m_run = -3e38f, l_run = 0.f;

  if (wv == 0) stage_tiles(smem0, 0, Kbank, VbankT, bh, 0);   // prologue

  for (int jb = 0; jb < 1024; jb += 32) {
    int slot = (jb >> 5) & 1;
    if (wv == 0) {
      if (jb + 32 < 1024) {
        stage_tiles(smem0, slot ^ 1, Kbank, VbankT, bh, jb + 32);
        __builtin_amdgcn_s_wait_tensorcnt(10);  // retire current slot's 10
      } else {
        __builtin_amdgcn_s_wait_tensorcnt(0);
      }
    }
    __syncthreads();
    const bf16_t* sBase = (const bf16_t*)(smem_raw + slot * 40960);

    float mj = (float)maskA[b * 1024 + jb + lane];
    v8f St0 = {}, St1 = {};
#pragma unroll
    for (int bi = 0; bi < 5; ++bi) {
      v16bf bq0 = (bsel_i == bi) ? qf[0] : zf;   // mask Q columns by bank
      v16bf bq1 = (bsel_i == bi) ? qf[1] : zf;
      const bf16_t* sK = sBase + bi * 2048;      // [32][64] tile in LDS
      St0 = wmma_bf16(load_a_frag(sK, 64, lane), bq0, St0);
      St0 = wmma_bf16(load_a_frag(sK + 32, 64, lane), bq1, St0);
      St1 = wmma_bf16(load_a_frag(sK + 16 * 64, 64, lane), bq0, St1);
      St1 = wmma_bf16(load_a_frag(sK + 16 * 64 + 32, 64, lane), bq1, St1);
    }
    // scale + mask (row j index of element e is jb + js*16 + e + hi8)
    float s0[8], s1[8];
#pragma unroll
    for (int e = 0; e < 8; ++e) {
      float mj0 = __shfl(mj, e + hi8, 32);
      float mj1 = __shfl(mj, 16 + e + hi8, 32);
      s0[e] = (mask_i && mj0 > 0.5f) ? St0[e] * 0.125f : -1e30f;
      s1[e] = (mask_i && mj1 > 0.5f) ? St1[e] * 0.125f : -1e30f;
    }
    // online softmax (stats per query column i == per lane, halves synced)
    float tmax = -3e38f;
#pragma unroll
    for (int e = 0; e < 8; ++e) tmax = fmaxf(tmax, fmaxf(s0[e], s1[e]));
    tmax = fmaxf(tmax, __shfl_xor(tmax, 16, 32));
    float m_new = fmaxf(m_run, tmax);
    float f = __expf(m_run - m_new);
    float p0[8], p1[8], ls = 0.f;
#pragma unroll
    for (int e = 0; e < 8; ++e) {
      p0[e] = __expf(s0[e] - m_new);
      p1[e] = __expf(s1[e] - m_new);
      ls += p0[e] + p1[e];
    }
    l_run = l_run * f + ls;
    m_run = m_new;
#pragma unroll
    for (int e = 0; e < 8; ++e) { O0[e] *= f; O1[e] *= f; O2[e] *= f; O3[e] *= f; }
    // build P^T B-fragment (K = j 0..31 rel jb) from the two C sub-tiles
    float o0x[8], o1x[8];
#pragma unroll
    for (int e = 0; e < 8; ++e) {
      o0x[e] = __shfl_xor(p0[e], 16, 32);
      o1x[e] = __shfl_xor(p1[e], 16, 32);
    }
    bool lo = lane < 16;
    v16bf pB;
#pragma unroll
    for (int e = 0; e < 8; ++e) pB[e] = (bf16_t)(lo ? p0[e] : o1x[e]);
#pragma unroll
    for (int e = 0; e < 8; ++e) pB[8 + e] = (bf16_t)(lo ? o0x[e] : p1[e]);
    // PV: A = V'^T rows D (from LDS), k-run over j; mask P columns per bank
#pragma unroll
    for (int bi = 0; bi < 5; ++bi) {
      v16bf pm = (bsel_i == bi) ? pB : zf;
      const bf16_t* sV = sBase + 10240 + bi * 2048;   // [64][32] tile in LDS
      O0 = wmma_bf16(load_a_frag(sV + 0 * 16 * 32, 32, lane), pm, O0);
      O1 = wmma_bf16(load_a_frag(sV + 1 * 16 * 32, 32, lane), pm, O1);
      O2 = wmma_bf16(load_a_frag(sV + 2 * 16 * 32, 32, lane), pm, O2);
      O3 = wmma_bf16(load_a_frag(sV + 3 * 16 * 32, 32, lane), pm, O3);
    }
    __syncthreads();   // slot reusable only after every wave is done with it
  }
  float l_tot = l_run + __shfl_xor(l_run, 16, 32);
  float inv = (l_tot > 0.f) ? 1.f / l_tot : 0.f;
  int i = i0 + (lane & 15);
  float* orow = out + ((size_t)b * 1024 + i) * 512 + h * 64;
#pragma unroll
  for (int e = 0; e < 8; ++e) {
    int D = e + hi8;
    orow[0 + D]  = O0[e] * inv;
    orow[16 + D] = O1[e] * inv;
    orow[32 + D] = O2[e] * inv;
    orow[48 + D] = O3[e] * inv;
  }
}

// ---------------- host launch ----------------------------------------------
extern "C" void kernel_launch(void* const* d_in, const int* in_sizes, int n_in,
                              void* d_out, int out_size, void* d_ws,
                              size_t ws_size, hipStream_t stream) {
  const float* query = (const float*)d_in[0];
  const float* key_t = (const float*)d_in[1];
  const float* value = (const float*)d_in[2];
  const int* b_seq = (const int*)d_in[3];
  const unsigned char* mask = (const unsigned char*)d_in[4];
  const float* lin = (const float*)d_in[5];
  const float* W1 = (const float*)d_in[6];
  const float* a1 = (const float*)d_in[7];
  const float* W2 = (const float*)d_in[8];
  const float* a2 = (const float*)d_in[9];
  float* out = (float*)d_out;

  char* w = (char*)d_ws;
  size_t off = 0;
  bf16_t* Wt = (bf16_t*)(w + off);        off += (size_t)3 * 5 * 512 * 512 * 2;
  bf16_t* W1m = (bf16_t*)(w + off);       off += (size_t)CCC * HH * 4096 * 2;
  bf16_t* W2mT = (bf16_t*)(w + off);      off += (size_t)CCC * HH * 4096 * 2;
  bf16_t* qBuf = (bf16_t*)(w + off);      off += (size_t)BBB * HH * NNN * DKK * 2;
  bf16_t* kBuf = (bf16_t*)(w + off);      off += (size_t)BBB * HH * NNN * DKK * 2;
  bf16_t* vBuf = (bf16_t*)(w + off);      off += (size_t)BBB * HH * NNN * DKK * 2;
  bf16_t* Kbank = (bf16_t*)(w + off);     off += (size_t)5 * BBB * HH * NNN * DKK * 2;
  bf16_t* VbankT = (bf16_t*)(w + off);    off += (size_t)5 * BBB * HH * NNN * DKK * 2;
  (void)ws_size; (void)in_sizes; (void)n_in; (void)out_size;

  moba_prep_weights<<<(3 * 5 * 512 * 512 + 255) / 256, 256, 0, stream>>>(lin, Wt);
  moba_mix<<<(2 * CCC * HH * 4096 + 255) / 256, 256, 0, stream>>>(W1, a1, W2, a2,
                                                                  W1m, W2mT);
  moba_proj<<<dim3(128, 8, 3), 128, 0, stream>>>(query, key_t, value, Wt, b_seq,
                                                 qBuf, kBuf, vBuf);
  moba_banks<<<dim3(64, 80, 2), 128, 0, stream>>>(kBuf, vBuf, b_seq, W1m, W2mT,
                                                  Kbank, VbankT);
  moba_attn<<<dim3(16, 16), 128, 81920, stream>>>(qBuf, Kbank, VbankT, b_seq,
                                                  mask, out);
}
